// GKNnet_77549929496729
// MI455X (gfx1250) — matrix operand order repaired and checked by reading, not compiled, not run on hardware
//
#include <hip/hip_runtime.h>
#include <hip/hip_bf16.h>

typedef __bf16 v16bf __attribute__((ext_vector_type(16)));
typedef float  v8f   __attribute__((ext_vector_type(8)));

#define HDIM   128
#define NNODES 50000
#define NEDGES 800000
#define KKNOTS 5
#define CTILES (HDIM / 16)     // 8 column tiles
#define KTILES (HDIM / 32)     // 4 k tiles

__device__ __forceinline__ unsigned short f2bf(float f) {
    unsigned int u = __builtin_bit_cast(unsigned int, f);
    u += 0x7FFFu + ((u >> 16) & 1u);           // round-to-nearest-even
    return (unsigned short)(u >> 16);
}

// ---------------------------------------------------------------------------
// float32 -> bf16 elementwise (row-major activations: A operand of the GEMMs)
__global__ void cvt_f32_bf16(const float* __restrict__ in,
                             unsigned short* __restrict__ out, long long n) {
    long long i = (long long)blockIdx.x * blockDim.x + threadIdx.x;
    if (i < n) out[i] = f2bf(in[i]);
}

__global__ void zero_f32(float* __restrict__ p, long long n) {
    long long i = (long long)blockIdx.x * blockDim.x + threadIdx.x;
    if (i < n) p[i] = 0.0f;
}

// ---------------------------------------------------------------------------
// Pack a 128x128 f32 weight matrix into per-lane WMMA B-fragment order (bf16):
//   dword index = ((kt*CTILES + ct)*32 + lane)*8 + i
//   holds B[kt*32 + (lane>>4)*16 + 2i, ct*16 + (lane&15)] and the k+1 half.
// transposeSrc=1 reads src as (Ncols x Kdim) row-major, i.e. B = src^T.
__global__ void pack_b_frag(const float* __restrict__ in,
                            unsigned int* __restrict__ outp, int transposeSrc) {
    int idx = blockIdx.x * blockDim.x + threadIdx.x;
    const int total = KTILES * CTILES * 32 * 8;
    if (idx >= total) return;
    int i    = idx & 7;
    int lane = (idx >> 3) & 31;
    int rest = idx >> 8;
    int ct   = rest % CTILES;
    int kt   = rest / CTILES;
    int col  = ct * 16 + (lane & 15);
    int k    = kt * 32 + (lane >> 4) * 16 + 2 * i;
    float lo, hi;
    if (transposeSrc) {
        lo = in[(size_t)col * HDIM + k];
        hi = in[(size_t)col * HDIM + k + 1];
    } else {
        lo = in[(size_t)k * HDIM + col];
        hi = in[(size_t)(k + 1) * HDIM + col];
    }
    outp[idx] = (unsigned int)f2bf(lo) | ((unsigned int)f2bf(hi) << 16);
}

// ---------------------------------------------------------------------------
// WMMA bf16 GEMM: C(M x 128) = A(M x 128) * B(128 x 128), f32 accumulate.
// A row-major bf16; B pre-packed in fragment order (2x b128 per k-step).
// One wave32 per 16x16 tile, 4 waves/block, K fully unrolled (4 WMMAs).
// MODE 0: outF = C            (relation transform)
// MODE 1: outB = bf16(relu(C + bias))          (hidden layers)
// MODE 2: outF = alpha*(C + bias) + beta       (KAN pre-activation)
template <int MODE>
__global__ __launch_bounds__(128) void wmma_gemm_bf16(
    const unsigned short* __restrict__ A, const unsigned int* __restrict__ Bp,
    float* __restrict__ outF, unsigned short* __restrict__ outB,
    const float* __restrict__ bias, const float* __restrict__ alpha,
    const float* __restrict__ beta, int M) {
    const int lane = threadIdx.x & 31;
    const int wave = threadIdx.x >> 5;
    const int tile = blockIdx.x * 4 + wave;
    if (tile >= (M >> 4) * CTILES) return;             // wave-uniform exit
    const int rowTile = tile / CTILES;
    const int colTile = tile % CTILES;

    union Frag { v16bf v; unsigned int u[8]; };
    Frag au, bu;
    v8f acc = {};

    const int arow = rowTile * 16 + (lane & 15);
    const int kg   = (lane >> 4) * 16;                 // A K-half: contiguous 16 per lane half
    const unsigned int* Arow = (const unsigned int*)(A + (size_t)arow * HDIM);
    const unsigned int* Bl   = Bp + ((size_t)colTile * 32 + lane) * 8;

#pragma unroll
    for (int kt = 0; kt < KTILES; ++kt) {
        const int k0 = kt * 32;
#pragma unroll
        for (int i = 0; i < 4; ++i) {                  // merges to 2x b128
            au.u[i]     = Arow[(k0 + kg + 2 * i) >> 1];
            au.u[i + 4] = Arow[(k0 + 8 + kg + 2 * i) >> 1];
        }
        const unsigned int* bptr = Bl + (size_t)kt * CTILES * 32 * 8;
#pragma unroll
        for (int i = 0; i < 8; ++i) bu.u[i] = bptr[i]; // merges to 2x b128
        acc = __builtin_amdgcn_wmma_f32_16x16x32_bf16(
            false, au.v, false, bu.v, (short)0, acc, false, false);
    }

    const int r0  = rowTile * 16 + (lane >> 4) * 8;    // C layout: VGPR v -> M = v(+8)
    const int col = colTile * 16 + (lane & 15);
    float bcol = 0.f, acol = 0.f, gcol = 0.f;
    if (MODE != 0) bcol = bias[col];
    if (MODE == 2) { acol = alpha[col]; gcol = beta[col]; }
#pragma unroll
    for (int v = 0; v < 8; ++v) {
        float val = acc[v];
        size_t off = (size_t)(r0 + v) * HDIM + col;
        if (MODE == 0) {
            outF[off] = val;
        } else if (MODE == 1) {
            outB[off] = f2bf(fmaxf(val + bcol, 0.0f));
        } else {
            outF[off] = acol * (val + bcol) + gcol;
        }
    }
}

// ---------------------------------------------------------------------------
// agg[row] += T[col] : one wave per edge, float4 gather + 4 atomic f32 adds
__global__ void scatter_add(const float* __restrict__ T,
                            const int* __restrict__ rows,
                            const int* __restrict__ cols,
                            float* __restrict__ agg, int E) {
    long long t = (long long)blockIdx.x * blockDim.x + threadIdx.x;
    int lane = (int)(t & 31);
    long long e = t >> 5;
    if (e >= E) return;
    int src = cols[e], dst = rows[e];
    float4 v = ((const float4*)(T + (size_t)src * HDIM))[lane];
    float* d = agg + (size_t)dst * HDIM + lane * 4;
    atomicAdd(d + 0, v.x);
    atomicAdd(d + 1, v.y);
    atomicAdd(d + 2, v.z);
    atomicAdd(d + 3, v.w);
}

// ---------------------------------------------------------------------------
// Natural cubic spline: solve 3x3 tridiagonal per feature (Thomas), M[0]=M[4]=0
__global__ void spline_solve(const float* __restrict__ knots,
                             const float* __restrict__ vals,   // H x K
                             float* __restrict__ Msp) {        // H x K
    int f = threadIdx.x;
    if (f >= HDIM) return;
    float t[KKNOTS], y[KKNOTS];
#pragma unroll
    for (int i = 0; i < KKNOTS; ++i) { t[i] = knots[i]; y[i] = vals[f * KKNOTS + i]; }
    float h[4], s[4];
#pragma unroll
    for (int i = 0; i < 4; ++i) { h[i] = t[i + 1] - t[i]; s[i] = (y[i + 1] - y[i]) / h[i]; }
    float r0 = 6.f * (s[1] - s[0]), r1 = 6.f * (s[2] - s[1]), r2 = 6.f * (s[3] - s[2]);
    float d0 = 2.f * (h[0] + h[1]), d1 = 2.f * (h[1] + h[2]), d2 = 2.f * (h[2] + h[3]);
    float u01 = h[1], u12 = h[2];
    float c0 = u01 / d0,            g0 = r0 / d0;
    float m1 = d1 - u01 * c0;
    float c1 = u12 / m1,            g1 = (r1 - u01 * g0) / m1;
    float m2 = d2 - u12 * c1;
    float g2 = (r2 - u12 * g1) / m2;
    float M2v = g2, M1v = g1 - c1 * M2v, M0v = g0 - c0 * M1v;
    Msp[f * KKNOTS + 0] = 0.f;  Msp[f * KKNOTS + 1] = M0v;
    Msp[f * KKNOTS + 2] = M1v;  Msp[f * KKNOTS + 3] = M2v;
    Msp[f * KKNOTS + 4] = 0.f;
}

__global__ void spline_eval(float* __restrict__ kd,
                            const float* __restrict__ knots,
                            const float* __restrict__ vals,
                            const float* __restrict__ Msp, long long total) {
    long long idx = (long long)blockIdx.x * blockDim.x + threadIdx.x;
    if (idx >= total) return;
    int f = (int)(idx & (HDIM - 1));
    float x = kd[idx];
    float t[KKNOTS];
#pragma unroll
    for (int i = 0; i < KKNOTS; ++i) t[i] = knots[i];
    int cnt = 0;
#pragma unroll
    for (int i = 0; i < KKNOTS; ++i) cnt += (x >= t[i]) ? 1 : 0;   // searchsorted 'right'
    int j = cnt - 1; j = j < 0 ? 0 : j; j = j > KKNOTS - 2 ? KKNOTS - 2 : j;
    float t0 = t[j], t1 = t[j + 1], hj = t1 - t0;
    float y0 = vals[f * KKNOTS + j], y1 = vals[f * KKNOTS + j + 1];
    float M0 = Msp[f * KKNOTS + j],  M1 = Msp[f * KKNOTS + j + 1];
    float a = t1 - x, b = x - t0;
    float inv6h = 1.f / (6.f * hj);
    kd[idx] = M0 * a * a * a * inv6h + M1 * b * b * b * inv6h
            + (y0 / hj - M0 * hj / 6.f) * a + (y1 / hj - M1 * hj / 6.f) * b;
}

// ---------------------------------------------------------------------------
// Final FC (C=1): one wave per node, float4 dot + shuffle reduce
__global__ void fc_kernel(const float* __restrict__ kan,
                          const float* __restrict__ fw,
                          const float* __restrict__ fb,
                          float* __restrict__ out, int Nn) {
    long long t = (long long)blockIdx.x * blockDim.x + threadIdx.x;
    int lane = (int)(t & 31);
    long long n = t >> 5;
    if (n >= Nn) return;
    float4 a = ((const float4*)(kan + (size_t)n * HDIM))[lane];
    float4 w = ((const float4*)fw)[lane];
    float s = a.x * w.x + a.y * w.y + a.z * w.z + a.w * w.w;
#pragma unroll
    for (int o = 16; o > 0; o >>= 1) s += __shfl_down(s, o, 32);
    if (lane == 0) out[n] = s + fb[0];
}

// ---------------------------------------------------------------------------
extern "C" void kernel_launch(void* const* d_in, const int* in_sizes, int n_in,
                              void* d_out, int out_size, void* d_ws, size_t ws_size,
                              hipStream_t stream) {
    (void)in_sizes; (void)n_in; (void)out_size; (void)ws_size;
    const float* x     = (const float*)d_in[0];
    const int*   e1    = (const int*)d_in[1];
    const int*   e2    = (const int*)d_in[2];
    const int*   e3    = (const int*)d_in[3];
    const float* W1    = (const float*)d_in[4];
    const float* l1w   = (const float*)d_in[5];
    const float* l1b   = (const float*)d_in[6];
    const float* W2    = (const float*)d_in[7];
    const float* l2w   = (const float*)d_in[8];
    const float* l2b   = (const float*)d_in[9];
    const float* kw    = (const float*)d_in[10];
    const float* kb    = (const float*)d_in[11];
    const float* alpha = (const float*)d_in[12];
    const float* beta  = (const float*)d_in[13];
    const float* knots = (const float*)d_in[14];
    const float* sv    = (const float*)d_in[15];
    const float* fw    = (const float*)d_in[16];
    const float* fb    = (const float*)d_in[17];
    float* out = (float*)d_out;

    char* ws = (char*)d_ws;
    size_t off = 0;
    auto alloc = [&](size_t b) { size_t p = off; off += (b + 255) & ~(size_t)255; return p; };
    const long long nx = (long long)NNODES * HDIM;
    const int PACKB = KTILES * CTILES * 32 * 8;                   // dwords per packed B
    unsigned short* xb  = (unsigned short*)(ws + alloc(nx * 2));  // bf16 GEMM A operand
    unsigned short* hb  = (unsigned short*)(ws + alloc(nx * 2));  // bf16 h1/h2
    float* T    = (float*)(ws + alloc(nx * 4));                   // per-relation x@W_r
    float* agg  = (float*)(ws + alloc(nx * 4));                   // scatter accumulator
    float* kpre = (float*)(ws + alloc(nx * 4));                   // KAN pre/post spline
    unsigned int* W1p = (unsigned int*)(ws + alloc((size_t)3 * PACKB * 4));
    unsigned int* W2p = (unsigned int*)(ws + alloc((size_t)3 * PACKB * 4));
    unsigned int* l1p = (unsigned int*)(ws + alloc((size_t)PACKB * 4));
    unsigned int* l2p = (unsigned int*)(ws + alloc((size_t)PACKB * 4));
    unsigned int* kwp = (unsigned int*)(ws + alloc((size_t)PACKB * 4));
    float* Msp = (float*)(ws + alloc((size_t)HDIM * KKNOTS * 4));

    const int TB = 256;
    const int WW = HDIM * HDIM;
    const int pb = (PACKB + TB - 1) / TB;

    // weight prep: pack all B operands into WMMA fragment order
    cvt_f32_bf16<<<(int)((nx + TB - 1) / TB), TB, 0, stream>>>(x, xb, nx);
    for (int r = 0; r < 3; ++r) {
        pack_b_frag<<<pb, TB, 0, stream>>>(W1 + (size_t)r * WW, W1p + (size_t)r * PACKB, 0);
        pack_b_frag<<<pb, TB, 0, stream>>>(W2 + (size_t)r * WW, W2p + (size_t)r * PACKB, 0);
    }
    pack_b_frag<<<pb, TB, 0, stream>>>(l1w, l1p, 1);   // B = lin_w^T
    pack_b_frag<<<pb, TB, 0, stream>>>(l2w, l2p, 1);
    pack_b_frag<<<pb, TB, 0, stream>>>(kw, kwp, 1);    // B = kan_w^T
    spline_solve<<<1, HDIM, 0, stream>>>(knots, sv, Msp);

    const int gemmBlocks = (NNODES / 16) * CTILES / 4; // 4 waves/block
    const long long st = (long long)NEDGES * 32;
    const int* ei[3] = {e1, e2, e3};

    // ---- layer 1 ----
    zero_f32<<<(int)((nx + TB - 1) / TB), TB, 0, stream>>>(agg, nx);
    for (int r = 0; r < 3; ++r) {
        wmma_gemm_bf16<0><<<gemmBlocks, 128, 0, stream>>>(
            xb, W1p + (size_t)r * PACKB, T, nullptr, nullptr, nullptr, nullptr, NNODES);
        scatter_add<<<(int)((st + TB - 1) / TB), TB, 0, stream>>>(
            T, ei[r], ei[r] + NEDGES, agg, NEDGES);
    }
    cvt_f32_bf16<<<(int)((nx + TB - 1) / TB), TB, 0, stream>>>(agg, xb, nx);
    wmma_gemm_bf16<1><<<gemmBlocks, 128, 0, stream>>>(            // h1 = relu(agg@l1w.T+b)
        xb, l1p, nullptr, hb, l1b, nullptr, nullptr, NNODES);

    // ---- layer 2 ----
    zero_f32<<<(int)((nx + TB - 1) / TB), TB, 0, stream>>>(agg, nx);
    for (int r = 0; r < 3; ++r) {
        wmma_gemm_bf16<0><<<gemmBlocks, 128, 0, stream>>>(
            hb, W2p + (size_t)r * PACKB, T, nullptr, nullptr, nullptr, nullptr, NNODES);
        scatter_add<<<(int)((st + TB - 1) / TB), TB, 0, stream>>>(
            T, ei[r], ei[r] + NEDGES, agg, NEDGES);
    }
    cvt_f32_bf16<<<(int)((nx + TB - 1) / TB), TB, 0, stream>>>(agg, xb, nx);
    wmma_gemm_bf16<1><<<gemmBlocks, 128, 0, stream>>>(            // h2 = relu(agg@l2w.T+b)
        xb, l2p, nullptr, hb, l2b, nullptr, nullptr, NNODES);

    // ---- KAN: k = alpha*(h2@kan_w.T + kan_b) + beta, then spline eval ----
    wmma_gemm_bf16<2><<<gemmBlocks, 128, 0, stream>>>(
        hb, kwp, kpre, nullptr, kb, alpha, beta, NNODES);
    spline_eval<<<(int)((nx + TB - 1) / TB), TB, 0, stream>>>(kpre, knots, sv, Msp, nx);

    // ---- final FC (C=1) ----
    fc_kernel<<<(int)(((long long)NNODES * 32 + TB - 1) / TB), TB, 0, stream>>>(
        kpre, fw, fb, out, NNODES);
}
